// CraftaxAgent_79989470920752
// MI455X (gfx1250) — matrix operand order repaired and checked by reading, not compiled, not run on hardware
//
#include <hip/hip_runtime.h>
#include <hip/hip_bf16.h>

// ---------------------------------------------------------------------------
// CraftaxAgent forward on MI455X (gfx1250, wave32).
//  - All GEMMs via v_wmma_f32_16x16x32_bf16 (16x16 D tiles, K=32 steps).
//  - Encoder L1 is HBM-bound (1.08 GB fp32 x): float4 loads, register-
//    pipelined LDS staging so global loads overlap WMMA; native bf16 cvt.
//  - The uniform 1/255 scale is folded into the (tiny) W1 staging instead of
//    the huge x stream: (x/255)·W1^T == x·(W1/255)^T.
//  - bf16-input GEMMs stage their A tiles with global_load_async_to_lds_b128
//    (ASYNCcnt) -- pure byte copies, no VGPR round trip.
//  - LSTM scan: fp32 VALU, one block per batch element (rows independent).
// ---------------------------------------------------------------------------

typedef __attribute__((ext_vector_type(16))) __bf16 v16bf;
typedef __attribute__((ext_vector_type(8)))  float  v8f;
typedef __attribute__((ext_vector_type(2)))  __bf16 bf16x2;
typedef unsigned short u16;

#define TB   32768   // T*B rows
#define TT   128     // T
#define BB   256     // B
#define HH   32      // H

static __device__ __forceinline__ u16 f32_to_bf16_bits(float f) {
    return __builtin_bit_cast(u16, (__bf16)f);      // native cvt, RNE
}
static __device__ __forceinline__ unsigned pack_bf16x2(float a, float b) {
    bf16x2 t;
    t[0] = (__bf16)a;                               // -> v_cvt_pk_bf16_f32
    t[1] = (__bf16)b;
    return __builtin_bit_cast(unsigned, t);
}
static __device__ __forceinline__ __bf16 bf16_from_bits(u16 h) {
    return __builtin_bit_cast(__bf16, h);
}
static __device__ __forceinline__ float sigmoidf_(float x) {
    return 1.0f / (1.0f + __expf(-x));
}

// ---------------------------------------------------------------------------
// Generic tiled GEMM:  out[M x n_store] = act(A[M x K] * (wscale*W)[N x K]^T + bias)
//   NT      : number of 16-wide N tiles (N = 16*NT)
//   AF32    : A is fp32 (pipelined float4 staging)
//             else A is bf16 bits (async b128 copy straight into LDS)
//   TANH    : apply tanh epilogue
//   OUTBF16 : store bf16 bits else fp32
// Block: 256 threads (8 waves); wave w owns rows 16w..16w+15 x all N tiles.
// Grid : M/128 blocks. All K-tile edges are multiples of 4 floats, so every
// float4 staging load is either fully valid or fully zero (no scalar path).
// ---------------------------------------------------------------------------
template<int NT, bool AF32, bool TANH, bool OUTBF16>
__global__ __launch_bounds__(256)
void gemm_wmma(const void* __restrict__ A, int lda,
               const float* __restrict__ W,        // [N x K] row-major
               const float* __restrict__ bias1,
               const float* __restrict__ bias2,    // optional (nullptr)
               void* __restrict__ out, int out_ld, int n_store,
               int K, float wscale)
{
    __shared__ u16 sA[128][32];       // A tile: 128 rows x 32 K (bf16 bits)
    __shared__ u16 sB[NT * 16][32];   // W tile: [n][k] (bf16 bits)
    constexpr int BV = (NT * 128 + 255) / 256;   // B float4 loads per thread

    const int tid  = threadIdx.x;
    const int lane = tid & 31;
    const int wave = tid >> 5;
    const int half = lane >> 4;       // K-half selector: this code packs k = 16*half + s
    const int l15  = lane & 15;
    const int rowBase = blockIdx.x * 128;

    const float* __restrict__ Af = (const float*)A;
    const u16*   __restrict__ Ab = (const u16*)A;

    v8f acc[NT];
    v8f zero = {};
#pragma unroll
    for (int i = 0; i < NT; i++) acc[i] = zero;

    // ---- fragment build + WMMA on the staged LDS tiles ----
    auto do_wmma = [&]() {
        v16bf afrag;
#pragma unroll
        for (int v = 0; v < 8; v++) {
            int kb = half * 16 + 2 * v;
            unsigned w = *(const unsigned*)&sA[wave * 16 + l15][kb];
            afrag[2 * v]     = bf16_from_bits((u16)w);
            afrag[2 * v + 1] = bf16_from_bits((u16)(w >> 16));
        }
#pragma unroll
        for (int nt = 0; nt < NT; nt++) {
            v16bf bfrag;
#pragma unroll
            for (int v = 0; v < 8; v++) {
                int kb = half * 16 + 2 * v;
                unsigned w = *(const unsigned*)&sB[nt * 16 + l15][kb];
                bfrag[2 * v]     = bf16_from_bits((u16)w);
                bfrag[2 * v + 1] = bf16_from_bits((u16)(w >> 16));
            }
            acc[nt] = __builtin_amdgcn_wmma_f32_16x16x32_bf16(
                false, afrag, false, bfrag, (short)0, acc[nt], false, false);
        }
    };

    if constexpr (AF32) {
        // ---- fp32 source: float4 loads, register-pipelined staging ----
        float4 ar[4];
        float4 br[BV];
        auto load_tile = [&](int kt) {
#pragma unroll
            for (int j = 0; j < 4; j++) {           // A: 1024 float4 / block
                int i = tid + j * 256;
                int r = i >> 3, kv = (i & 7) * 4;
                int gk = kt + kv;
                float4 v{0.f, 0.f, 0.f, 0.f};
                if (gk + 3 < K)
                    v = *(const float4*)(Af + (size_t)(rowBase + r) * lda + gk);
                ar[j] = v;
            }
#pragma unroll
            for (int j = 0; j < BV; j++) {          // B: NT*128 float4 / block
                int i = tid + j * 256;
                float4 v{0.f, 0.f, 0.f, 0.f};
                if (i < NT * 128) {
                    int n = i >> 3, kv = (i & 7) * 4;
                    int gk = kt + kv;
                    if (gk + 3 < K)
                        v = *(const float4*)(W + (size_t)n * K + gk);
                }
                br[j] = v;
            }
        };
        auto store_tile = [&]() {
#pragma unroll
            for (int j = 0; j < 4; j++) {
                int i = tid + j * 256;
                int r = i >> 3, kv = (i & 7) * 4;
                float4 v = ar[j];
                uint2 pk;
                pk.x = pack_bf16x2(v.x, v.y);       // pure convert, no scale
                pk.y = pack_bf16x2(v.z, v.w);
                *(uint2*)&sA[r][kv] = pk;           // ds_store_b64
            }
#pragma unroll
            for (int j = 0; j < BV; j++) {
                int i = tid + j * 256;
                if (i < NT * 128) {
                    int n = i >> 3, kv = (i & 7) * 4;
                    float4 v = br[j];
                    uint2 pk;
                    pk.x = pack_bf16x2(v.x * wscale, v.y * wscale);
                    pk.y = pack_bf16x2(v.z * wscale, v.w * wscale);
                    *(uint2*)&sB[n][kv] = pk;
                }
            }
        };

        load_tile(0);
        for (int kt = 0; kt < K; kt += 32) {
            store_tile();
            __syncthreads();
            if (kt + 32 < K) {
                load_tile(kt + 32);                 // overlaps WMMA below
                __builtin_prefetch(&Af[(size_t)(rowBase + (tid & 127)) * lda + kt + 64], 0, 1);
            }
            do_wmma();
            __syncthreads();
        }
    } else {
        // ---- bf16 source: async copy A tile straight into LDS ----
        for (int kt = 0; kt < K; kt += 32) {
#pragma unroll
            for (int j = 0; j < 2; j++) {           // 512 x 16B chunks / block
                int i = tid + j * 256;
                int r = i >> 2, c = (i & 3) * 8;    // 8 u16 = 16 bytes
                unsigned ldsoff = (unsigned)(size_t)&sA[r][c];
                unsigned long long ga =
                    (unsigned long long)(size_t)(Ab + (size_t)(rowBase + r) * lda + kt + c);
                asm volatile("global_load_async_to_lds_b128 %0, %1, off"
                             :: "v"(ldsoff), "v"(ga) : "memory");
            }
            // B tile: fp32 W -> bf16 LDS (tiny)
            for (int i = tid; i < NT * 128; i += 256) {
                int n = i >> 3, kv = (i & 7) * 4;
                int gk = kt + kv;
                float4 v{0.f, 0.f, 0.f, 0.f};
                if (gk + 3 < K)
                    v = *(const float4*)(W + (size_t)n * K + gk);
                uint2 pk;
                pk.x = pack_bf16x2(v.x * wscale, v.y * wscale);
                pk.y = pack_bf16x2(v.z * wscale, v.w * wscale);
                *(uint2*)&sB[n][kv] = pk;
            }
            asm volatile("s_wait_asynccnt 0x0" ::: "memory");
            __syncthreads();
            do_wmma();
            __syncthreads();
        }
    }

    // ---- epilogue: bias (+optional second bias), activation, store ----
#pragma unroll
    for (int nt = 0; nt < NT; nt++) {
        int col = nt * 16 + l15;
        float bv = 0.0f;
        if (col < n_store) {
            bv = bias1[col];
            if (bias2) bv += bias2[col];
        }
#pragma unroll
        for (int r = 0; r < 8; r++) {
            int row16 = r + 8 * half;            // C/D layout: lanes 16-31 -> M+8
            int grow  = rowBase + wave * 16 + row16;
            if (col < n_store) {
                float v = acc[nt][r] + bv;
                if constexpr (TANH) v = tanhf(v);
                if constexpr (OUTBF16)
                    ((u16*)out)[(size_t)grow * out_ld + col] = f32_to_bf16_bits(v);
                else
                    ((float*)out)[(size_t)grow * out_ld + col] = v;
            }
        }
    }
}

// ---------------------------------------------------------------------------
// LSTM scan. One block per batch element (batch rows are independent across
// the scan); 128 threads = one thread per gate. State h lives in LDS, c in
// registers of threads 0..31. Gate order i,f,g,o (torch convention).
// ---------------------------------------------------------------------------
__global__ __launch_bounds__(128)
void lstm_kernel(const float* __restrict__ xg,    // [T*B x 128]
                 const float* __restrict__ done,  // [T*B]
                 const float* __restrict__ Whh,   // [128 x 32]
                 const float* __restrict__ h0,    // [B x 32]
                 const float* __restrict__ c0,    // [B x 32]
                 float* __restrict__ feat,        // [T*B x 32]
                 float* __restrict__ outHN,       // [B x 32]
                 float* __restrict__ outCN)       // [B x 32]
{
    const int b = blockIdx.x;      // batch element
    const int g = threadIdx.x;     // gate index 0..127

    __shared__ float hs[HH];
    __shared__ float gates[4 * HH];

    float wr[HH];
#pragma unroll
    for (int k = 0; k < HH; k++) wr[k] = Whh[g * HH + k];

    float cs = 0.0f;
    if (g < HH) { hs[g] = h0[b * HH + g]; cs = c0[b * HH + g]; }
    __syncthreads();

    for (int t = 0; t < TT; t++) {
        const int row = t * BB + b;
        const float m = 1.0f - done[row];
        if (g < HH) hs[g] *= m;          // done-mask resets state
        cs *= m;
        __syncthreads();

        float acc = xg[(size_t)row * 128 + g];
#pragma unroll
        for (int k = 0; k < HH; k++) acc += wr[k] * hs[k];
        gates[g] = acc;
        __syncthreads();

        if (g < HH) {
            float ig = sigmoidf_(gates[g]);
            float fg = sigmoidf_(gates[HH + g]);
            float gg = tanhf(gates[2 * HH + g]);
            float og = sigmoidf_(gates[3 * HH + g]);
            cs = fg * cs + ig * gg;
            float h = og * tanhf(cs);
            hs[g] = h;
            feat[(size_t)row * HH + g] = h;
        }
        __syncthreads();
    }

    if (g < HH) {
        outHN[b * HH + g] = hs[g];
        outCN[b * HH + g] = cs;
    }
}

// ---------------------------------------------------------------------------
// Pack actor + critic heads into one [48 x 32] weight (rows 44..47 zero) so
// the head GEMM can run through WMMA with NT=3 and n_store=44.
// ---------------------------------------------------------------------------
__global__ void pack_heads(const float* __restrict__ Wa, const float* __restrict__ ba,
                           const float* __restrict__ Wc, const float* __restrict__ bc,
                           float* __restrict__ Wcomb, float* __restrict__ bcomb)
{
    int i = blockIdx.x * blockDim.x + threadIdx.x;
    if (i < 48 * 32) {
        int n = i / 32, k = i % 32;
        float v = (n < 43) ? Wa[n * 32 + k] : ((n == 43) ? Wc[k] : 0.0f);
        Wcomb[i] = v;
    }
    if (i < 48) bcomb[i] = (i < 43) ? ba[i] : ((i == 43) ? bc[0] : 0.0f);
}

// ---------------------------------------------------------------------------
extern "C" void kernel_launch(void* const* d_in, const int* in_sizes, int n_in,
                              void* d_out, int out_size, void* d_ws, size_t ws_size,
                              hipStream_t stream)
{
    (void)in_sizes; (void)n_in; (void)out_size; (void)ws_size;

    const float* x    = (const float*)d_in[0];
    const float* done = (const float*)d_in[1];
    const float* h0   = (const float*)d_in[2];
    const float* c0   = (const float*)d_in[3];
    const float* W1   = (const float*)d_in[4];
    const float* b1   = (const float*)d_in[5];
    const float* W2   = (const float*)d_in[6];
    const float* b2   = (const float*)d_in[7];
    const float* W3   = (const float*)d_in[8];
    const float* b3   = (const float*)d_in[9];
    const float* Wih  = (const float*)d_in[10];
    const float* Whh  = (const float*)d_in[11];
    const float* bih  = (const float*)d_in[12];
    const float* bhh  = (const float*)d_in[13];
    const float* Wa   = (const float*)d_in[14];
    const float* ba   = (const float*)d_in[15];
    const float* Wc   = (const float*)d_in[16];
    const float* bc   = (const float*)d_in[17];

    // workspace layout (256B aligned)
    char* w = (char*)d_ws;
    size_t off = 0;
    auto take = [&](size_t bytes) { size_t o = off; off += (bytes + 255) & ~(size_t)255; return o; };
    u16*   H1    = (u16*)  (w + take((size_t)TB * 64 * sizeof(u16)));
    u16*   H2    = (u16*)  (w + take((size_t)TB * 64 * sizeof(u16)));
    u16*   H3    = (u16*)  (w + take((size_t)TB * 32 * sizeof(u16)));
    float* XG    = (float*)(w + take((size_t)TB * 128 * sizeof(float)));
    float* FEAT  = (float*)(w + take((size_t)TB * 32 * sizeof(float)));
    float* WCOMB = (float*)(w + take((size_t)48 * 32 * sizeof(float)));
    float* BCOMB = (float*)(w + take((size_t)48 * sizeof(float)));

    float* out_f  = (float*)d_out;                 // [T*B x 44]
    float* out_hN = out_f + (size_t)TB * 44;       // [B x 32]
    float* out_cN = out_hN + (size_t)BB * HH;      // [B x 32]

    dim3 blk(256);
    dim3 grd(TB / 128);   // 256 blocks of 128 rows

    // Encoder: tanh(x·(W1/255)^T + b1) -> tanh(· W2^T + b2) -> tanh(· W3^T + b3)
    gemm_wmma<4, true,  true,  true ><<<grd, blk, 0, stream>>>(
        x, 8268, W1, b1, nullptr, H1, 64, 64, 8268, 1.0f / 255.0f);
    gemm_wmma<4, false, true,  true ><<<grd, blk, 0, stream>>>(
        H1, 64, W2, b2, nullptr, H2, 64, 64, 64, 1.0f);
    gemm_wmma<2, false, true,  true ><<<grd, blk, 0, stream>>>(
        H2, 64, W3, b3, nullptr, H3, 32, 32, 64, 1.0f);

    // Input gate projections for all steps: xg = H3 Wih^T + (bih + bhh)
    gemm_wmma<8, false, false, false><<<grd, blk, 0, stream>>>(
        H3, 32, Wih, bih, bhh, XG, 128, 128, 32, 1.0f);

    // Heads weight packing (only head GEMM depends on it)
    pack_heads<<<6, blk, 0, stream>>>(Wa, ba, Wc, bc, WCOMB, BCOMB);

    // Sequential LSTM scan
    lstm_kernel<<<BB, 128, 0, stream>>>(XG, done, Whh, h0, c0, FEAT, out_hN, out_cN);

    // Heads: [logits | value] = feat Wcomb^T + bcomb, store 44 of 48 cols
    gemm_wmma<3, true, false, false><<<grd, blk, 0, stream>>>(
        FEAT, 32, WCOMB, BCOMB, nullptr, out_f, 44, 44, 32, 1.0f);
}
